// MultiEmbeder_19834158973302
// MI455X (gfx1250) — matrix operand order repaired and checked by reading, not compiled, not run on hardware
//
#include <hip/hip_runtime.h>
#include <cstdint>
#include <cstddef>

#define H_DIM   512
#define BATCH   128
#define S_TOKC  512
#define S_DESCC 128
#define N_NODESC 65536
#define MARGINF 0.05f

typedef __bf16 bf16x16 __attribute__((ext_vector_type(16)));
typedef float  f32x8   __attribute__((ext_vector_type(8)));

union ABFrag { uint4 q[2]; bf16x16 v; };

__device__ __forceinline__ unsigned short f2bf(float f) {
  union { float f; unsigned int u; } x; x.f = f;
  unsigned int u = x.u;
  unsigned int r = u + 0x7fffu + ((u >> 16) & 1u);   // round-to-nearest-even
  if ((u & 0x7fffffffu) > 0x7f800000u) r = u | 0x00400000u;  // quiet NaN
  return (unsigned short)(r >> 16);
}

// ---------------------------------------------------------------------------
// K0: convert W [K,N] f32 (row-major, y = x @ W) -> WT [N,K] bf16 bits
// ---------------------------------------------------------------------------
__global__ void prep_wT(const float* __restrict__ W, unsigned short* __restrict__ WT,
                        int K, int N) {
  int idx = blockIdx.x * blockDim.x + threadIdx.x;
  if (idx >= K * N) return;
  int k = idx / N, n = idx - k * N;
  WT[(size_t)n * K + k] = f2bf(W[idx]);
}

// ---------------------------------------------------------------------------
// K1: fused GEMM. mode 0: logits[m] = sum_n tanh((X@W)[m,n] + b[n]) * v[n] + c
//     mode 1: out_mat[m,n] = tanh((X@W)[m,n] + b[n])      (N fixed = 512)
// Block = 128 threads (4 waves). Block owns 16 rows; wave w owns N-tiles
// w*128 + t*16, t=0..7. K-loop step 32 with bf16 WMMA 16x16x32.
// ---------------------------------------------------------------------------
__global__ __launch_bounds__(128) void gemm_attn(
    const float* __restrict__ X, int Kdim, int pre_tanh,
    const unsigned short* __restrict__ WT,
    const float* __restrict__ bias, const float* __restrict__ vvec,
    const float* __restrict__ cptr,
    float* __restrict__ out_logits, float* __restrict__ out_mat, int mode)
{
  __shared__ unsigned short a_lds[16 * 40];   // 16 rows, 32 used cols, pad to 40
  __shared__ float rowsum[16];

  const int tid  = threadIdx.x;
  const int lane = tid & 31;
  const int wave = tid >> 5;
  const int half = lane >> 4;
  const int ln   = lane & 15;
  const int row0 = blockIdx.x * 16;

  if (tid < 16) rowsum[tid] = 0.0f;

  f32x8 acc[8];
#pragma unroll
  for (int t = 0; t < 8; ++t)
#pragma unroll
    for (int r = 0; r < 8; ++r) acc[t][r] = 0.0f;

  for (int k0 = 0; k0 < Kdim; k0 += 32) {
    // --- stage A tile (16 rows x 32 K) into LDS as bf16, optional pre-tanh ---
    {
      int e  = tid * 4;          // 128 threads * 4 floats = 512 elements
      int ar = e >> 5;           // 0..15
      int ac = e & 31;           // 0..28 step 4
      const float4 f4 = *reinterpret_cast<const float4*>(
          X + (size_t)(row0 + ar) * Kdim + k0 + ac);
      float v0 = f4.x, v1 = f4.y, v2 = f4.z, v3 = f4.w;
      if (pre_tanh) { v0 = tanhf(v0); v1 = tanhf(v1); v2 = tanhf(v2); v3 = tanhf(v3); }
      unsigned short* dst = &a_lds[ar * 40 + ac];
      dst[0] = f2bf(v0); dst[1] = f2bf(v1); dst[2] = f2bf(v2); dst[3] = f2bf(v3);
    }
    __syncthreads();

    // --- A fragment: lane holds row m=ln; elem e -> K = (e&7) + 8*half + 16*(e>>3)
    ABFrag a;
    {
      const unsigned short* ap = &a_lds[ln * 40 + 8 * half];
      a.q[0] = *reinterpret_cast<const uint4*>(ap);        // K = k0 + 8*half .. +7
      a.q[1] = *reinterpret_cast<const uint4*>(ap + 16);   // K = k0+16+8*half .. +7
    }

#pragma unroll
    for (int t = 0; t < 8; ++t) {
      const int n = wave * 128 + t * 16 + ln;   // B column for this lane
      const unsigned short* bp = WT + (size_t)n * Kdim + k0 + 8 * half;
      ABFrag b;
      b.q[0] = *reinterpret_cast<const uint4*>(bp);        // K = k0 + 8*half .. +7
      b.q[1] = *reinterpret_cast<const uint4*>(bp + 16);   // K = k0+16+8*half .. +7
      acc[t] = __builtin_amdgcn_wmma_f32_16x16x32_bf16(
                 false, a.v, false, b.v, (short)0, acc[t], false, false);
    }
    __syncthreads();
  }

  if (mode == 0) {
    // epilogue: per-row dot of tanh(y+b) with v, reduced over all N=512
    float rsum[8];
#pragma unroll
    for (int r = 0; r < 8; ++r) rsum[r] = 0.0f;
#pragma unroll
    for (int t = 0; t < 8; ++t) {
      const int n  = wave * 128 + t * 16 + ln;
      const float bn = bias[n], vn = vvec[n];
#pragma unroll
      for (int r = 0; r < 8; ++r)
        rsum[r] += tanhf(acc[t][r] + bn) * vn;
    }
#pragma unroll
    for (int r = 0; r < 8; ++r) {
      rsum[r] += __shfl_xor(rsum[r], 1, 32);
      rsum[r] += __shfl_xor(rsum[r], 2, 32);
      rsum[r] += __shfl_xor(rsum[r], 4, 32);
      rsum[r] += __shfl_xor(rsum[r], 8, 32);
    }
    if (ln == 0) {
#pragma unroll
      for (int r = 0; r < 8; ++r)
        atomicAdd(&rowsum[8 * half + r], rsum[r]);   // C layout: row = r + 8*half
    }
    __syncthreads();
    if (tid < 16) {
      const float cval = cptr ? *cptr : 0.0f;
      out_logits[row0 + tid] = rowsum[tid] + cval;
    }
  } else {
    // epilogue: write tanh(y + b) matrix tile
#pragma unroll
    for (int t = 0; t < 8; ++t) {
      const int n = wave * 128 + t * 16 + ln;
      const float bn = bias[n];
#pragma unroll
      for (int r = 0; r < 8; ++r) {
        const int m = row0 + 8 * half + r;
        out_mat[(size_t)m * H_DIM + n] = tanhf(acc[t][r] + bn);
      }
    }
  }
}

// ---------------------------------------------------------------------------
// K2: masked softmax attention pooling + n_tanh output transform
// one block (256 thr) per batch row
// ---------------------------------------------------------------------------
__global__ __launch_bounds__(256) void masked_pool(
    const float* __restrict__ logits, const float* __restrict__ feat,
    const int* __restrict__ lens, float* __restrict__ out,
    int S, int n_tanh, int ostride, int ooff)
{
  const int b = blockIdx.x, tid = threadIdx.x;
  __shared__ float w[S_TOKC];
  __shared__ float red[256];
  const int len = lens[b];

  float lmax = -INFINITY;
  for (int s = tid; s < S; s += 256) {
    float l = (s < len) ? logits[b * S + s] : -INFINITY;
    w[s] = l;
    lmax = fmaxf(lmax, l);
  }
  red[tid] = lmax; __syncthreads();
  for (int off = 128; off > 0; off >>= 1) {
    if (tid < off) red[tid] = fmaxf(red[tid], red[tid + off]);
    __syncthreads();
  }
  lmax = red[0]; __syncthreads();

  float lsum = 0.0f;
  for (int s = tid; s < S; s += 256) {
    float e = (s < len) ? __expf(w[s] - lmax) : 0.0f;
    w[s] = e;
    lsum += e;
  }
  red[tid] = lsum; __syncthreads();
  for (int off = 128; off > 0; off >>= 1) {
    if (tid < off) red[tid] += red[tid + off];
    __syncthreads();
  }
  const float inv = 1.0f / red[0];
  __syncthreads();

  for (int h = tid; h < H_DIM; h += 256) {
    float acc = 0.0f;
    const float* fb = feat + ((size_t)b * S) * H_DIM + h;
    for (int s = 0; s < len; ++s)
      acc += w[s] * fb[(size_t)s * H_DIM];
    acc *= inv;
    for (int i = 0; i < n_tanh; ++i) acc = tanhf(acc);
    out[(size_t)b * ostride + ooff + h] = acc;
  }
}

// ---------------------------------------------------------------------------
// K3a: segment boundaries (seg_ids sorted) via binary search
// ---------------------------------------------------------------------------
__global__ void seg_bounds(const int* __restrict__ seg, int* __restrict__ start,
                           int N, int B) {
  int b = blockIdx.x * blockDim.x + threadIdx.x;
  if (b > B) return;
  if (b == B) { start[B] = N; return; }
  int lo = 0, hi = N;
  while (lo < hi) { int mid = (lo + hi) >> 1; if (seg[mid] < b) lo = mid + 1; else hi = mid; }
  start[b] = lo;
}

// ---------------------------------------------------------------------------
// K3b: per-segment softmax pool over tanh(node_h); out = tanh(pool)
// one block (256 thr) per segment
// ---------------------------------------------------------------------------
__global__ __launch_bounds__(256) void seg_pool(
    const float* __restrict__ logits, const float* __restrict__ node_h,
    const int* __restrict__ segstart, float* __restrict__ out,
    int ostride, int ooff)
{
  const int b = blockIdx.x, tid = threadIdx.x;
  const int s0 = segstart[b], s1 = segstart[b + 1];
  __shared__ float red[256];
  __shared__ float wbuf[2048];

  float lmax = -INFINITY;
  for (int i = s0 + tid; i < s1; i += 256) lmax = fmaxf(lmax, logits[i]);
  red[tid] = lmax; __syncthreads();
  for (int off = 128; off > 0; off >>= 1) {
    if (tid < off) red[tid] = fmaxf(red[tid], red[tid + off]);
    __syncthreads();
  }
  lmax = red[0]; __syncthreads();

  float lsum = 0.0f;
  for (int i = s0 + tid; i < s1; i += 256) lsum += __expf(logits[i] - lmax);
  red[tid] = lsum; __syncthreads();
  for (int off = 128; off > 0; off >>= 1) {
    if (tid < off) red[tid] += red[tid + off];
    __syncthreads();
  }
  const float inv = (s1 > s0) ? 1.0f / red[0] : 0.0f;
  __syncthreads();

  float acc0 = 0.0f, acc1 = 0.0f;
  for (int c0 = s0; c0 < s1; c0 += 2048) {
    const int cn = min(2048, s1 - c0);
    __syncthreads();
    for (int i = tid; i < cn; i += 256)
      wbuf[i] = __expf(logits[c0 + i] - lmax) * inv;
    __syncthreads();
    for (int i = 0; i < cn; ++i) {
      const float wv = wbuf[i];
      const float* row = node_h + (size_t)(c0 + i) * H_DIM;
      acc0 += wv * tanhf(row[tid]);
      acc1 += wv * tanhf(row[tid + 256]);
    }
  }
  out[(size_t)b * ostride + ooff + tid]       = tanhf(acc0);
  out[(size_t)b * ostride + ooff + tid + 256] = tanhf(acc1);
}

// ---------------------------------------------------------------------------
// K5: cosine sims + hinge loss (mean over batch into d_out[0])
// ---------------------------------------------------------------------------
__global__ void loss_init(float* out) { if (threadIdx.x == 0) out[0] = 0.0f; }

__device__ __forceinline__ float blk_sum(float v, float* red) {
  const int tid = threadIdx.x;
  red[tid] = v; __syncthreads();
  for (int off = 128; off > 0; off >>= 1) {
    if (tid < off) red[tid] += red[tid + off];
    __syncthreads();
  }
  float r = red[0]; __syncthreads();
  return r;
}

__global__ __launch_bounds__(256) void loss_kernel(
    const float* __restrict__ code, const float* __restrict__ da,
    const float* __restrict__ dn, float* __restrict__ out)
{
  const int b = blockIdx.x, tid = threadIdx.x;
  __shared__ float red[256];
  const float c0 = code[(size_t)b * H_DIM + tid];
  const float c1 = code[(size_t)b * H_DIM + tid + 256];
  const float a0 = da[(size_t)b * H_DIM + tid];
  const float a1 = da[(size_t)b * H_DIM + tid + 256];
  const float n0 = dn[(size_t)b * H_DIM + tid];
  const float n1 = dn[(size_t)b * H_DIM + tid + 256];

  const float scc = blk_sum(c0 * c0 + c1 * c1, red);
  const float saa = blk_sum(a0 * a0 + a1 * a1, red);
  const float snn = blk_sum(n0 * n0 + n1 * n1, red);
  const float sca = blk_sum(c0 * a0 + c1 * a1, red);
  const float scn = blk_sum(c0 * n0 + c1 * n1, red);

  if (tid == 0) {
    const float nc = fmaxf(sqrtf(scc), 1e-8f);
    const float na = fmaxf(sqrtf(saa), 1e-8f);
    const float nn = fmaxf(sqrtf(snn), 1e-8f);
    const float sim_a = sca / (nc * na);
    const float sim_n = scn / (nc * nn);
    const float term = fmaxf(MARGINF - sim_a + sim_n, 1e-6f);
    atomicAdd(out, term * (1.0f / (float)BATCH));
  }
}

// ---------------------------------------------------------------------------
// workspace layout (bytes)
// ---------------------------------------------------------------------------
static const size_t OFF_WTT  = 0;                         // 512*512*2
static const size_t OFF_WAT  = OFF_WTT  + 524288;
static const size_t OFF_WDT  = OFF_WAT  + 524288;
static const size_t OFF_WFT  = OFF_WDT  + 524288;         // 1024*512*2
static const size_t OFF_TOKL = OFF_WFT  + 1048576;        // 65536*4
static const size_t OFF_NODL = OFF_TOKL + 262144;         // 65536*4
static const size_t OFF_DAL  = OFF_NODL + 262144;         // 16384*4
static const size_t OFF_DNL  = OFF_DAL  + 65536;
static const size_t OFF_FUSE = OFF_DNL  + 65536;          // 128*1024*4
static const size_t OFF_DAP  = OFF_FUSE + 524288;         // 128*512*4
static const size_t OFF_DNP  = OFF_DAP  + 262144;
static const size_t OFF_CODE = OFF_DNP  + 262144;
static const size_t OFF_SEGS = OFF_CODE + 262144;         // (B+1)*4

extern "C" void kernel_launch(void* const* d_in, const int* in_sizes, int n_in,
                              void* d_out, int out_size, void* d_ws, size_t ws_size,
                              hipStream_t stream) {
  (void)in_sizes; (void)n_in; (void)out_size; (void)ws_size;

  const float* token_feat = (const float*)d_in[0];
  const int*   token_len  = (const int*)  d_in[1];
  const float* node_h     = (const float*)d_in[2];
  const int*   seg_ids    = (const int*)  d_in[3];
  const float* desc_a     = (const float*)d_in[4];
  const int*   desc_a_len = (const int*)  d_in[5];
  const float* desc_n     = (const float*)d_in[6];
  const int*   desc_n_len = (const int*)  d_in[7];
  const float* Wt = (const float*)d_in[8];
  const float* bt = (const float*)d_in[9];
  const float* vt = (const float*)d_in[10];
  const float* ct = (const float*)d_in[11];
  const float* Wa = (const float*)d_in[12];
  const float* ba = (const float*)d_in[13];
  const float* va = (const float*)d_in[14];
  const float* ca = (const float*)d_in[15];
  const float* Wd = (const float*)d_in[16];
  const float* bd = (const float*)d_in[17];
  const float* vd = (const float*)d_in[18];
  const float* cd = (const float*)d_in[19];
  const float* Wf = (const float*)d_in[20];
  const float* bfv = (const float*)d_in[21];

  char* ws = (char*)d_ws;
  unsigned short* WtT = (unsigned short*)(ws + OFF_WTT);
  unsigned short* WaT = (unsigned short*)(ws + OFF_WAT);
  unsigned short* WdT = (unsigned short*)(ws + OFF_WDT);
  unsigned short* WfT = (unsigned short*)(ws + OFF_WFT);
  float* tokL  = (float*)(ws + OFF_TOKL);
  float* nodL  = (float*)(ws + OFF_NODL);
  float* daL   = (float*)(ws + OFF_DAL);
  float* dnL   = (float*)(ws + OFF_DNL);
  float* fuse  = (float*)(ws + OFF_FUSE);
  float* daP   = (float*)(ws + OFF_DAP);
  float* dnP   = (float*)(ws + OFF_DNP);
  float* codeF = (float*)(ws + OFF_CODE);
  int*   segS  = (int*)  (ws + OFF_SEGS);
  float* loss  = (float*)d_out;

  // weight prep (bf16, [N,K])
  prep_wT<<<(512 * 512 + 255) / 256, 256, 0, stream>>>(Wt, WtT, 512, 512);
  prep_wT<<<(512 * 512 + 255) / 256, 256, 0, stream>>>(Wa, WaT, 512, 512);
  prep_wT<<<(512 * 512 + 255) / 256, 256, 0, stream>>>(Wd, WdT, 512, 512);
  prep_wT<<<(1024 * 512 + 255) / 256, 256, 0, stream>>>(Wf, WfT, 1024, 512);

  // attention logits (WMMA bf16 GEMMs, tanh·v epilogue)
  gemm_attn<<<(BATCH * S_TOKC) / 16, 128, 0, stream>>>(
      token_feat, 512, 0, WtT, bt, vt, ct, tokL, nullptr, 0);
  gemm_attn<<<N_NODESC / 16, 128, 0, stream>>>(
      node_h, 512, 1, WaT, ba, va, ca, nodL, nullptr, 0);  // pre-tanh on node states
  gemm_attn<<<(BATCH * S_DESCC) / 16, 128, 0, stream>>>(
      desc_a, 512, 0, WdT, bd, vd, cd, daL, nullptr, 0);
  gemm_attn<<<(BATCH * S_DESCC) / 16, 128, 0, stream>>>(
      desc_n, 512, 0, WdT, bd, vd, cd, dnL, nullptr, 0);

  // pooling
  masked_pool<<<BATCH, 256, 0, stream>>>(tokL, token_feat, token_len,
                                         fuse, S_TOKC, 1, 1024, 0);
  seg_bounds<<<1, 256, 0, stream>>>(seg_ids, segS, N_NODESC, BATCH);
  seg_pool<<<BATCH, 256, 0, stream>>>(nodL, node_h, segS, fuse, 1024, 512);
  masked_pool<<<BATCH, 256, 0, stream>>>(daL, desc_a, desc_a_len,
                                         daP, S_DESCC, 2, 512, 0);
  masked_pool<<<BATCH, 256, 0, stream>>>(dnL, desc_n, desc_n_len,
                                         dnP, S_DESCC, 2, 512, 0);

  // fuse: code_feat = tanh(concat(tok, ast) @ Wf + bf)   (WMMA, mode 1)
  gemm_attn<<<BATCH / 16, 128, 0, stream>>>(
      fuse, 1024, 0, WfT, bfv, nullptr, nullptr, nullptr, codeF, 1);

  // loss
  loss_init<<<1, 1, 0, stream>>>(loss);
  loss_kernel<<<BATCH, 256, 0, stream>>>(codeF, daP, dnP, loss);
}
